// MixtureOfAttentionHeads_21354577396447
// MI455X (gfx1250) — matrix operand order repaired and checked by reading, not compiled, not run on hardware
//
#include <hip/hip_runtime.h>
#include <cstdint>
#include <cstddef>

// ---------------- problem constants ----------------
#define B_   4
#define T_   2048
#define D_   1024
#define DH_  128
#define H_   2
#define G_   3
#define BT_  (B_ * T_)
#define WIN_ 128
#define MRP_ 32
#define RELV_ 65   // 2*MRP+1

typedef unsigned short u16_t;

typedef __attribute__((ext_vector_type(16))) __bf16 v16bf;
typedef __attribute__((ext_vector_type(8)))  __bf16 v8bf;
typedef __attribute__((ext_vector_type(8)))  float  v8f;

// ---- Tensor Data Mover availability ----
#if defined(__has_builtin)
#if __has_builtin(__builtin_amdgcn_tensor_load_to_lds) && \
    __has_builtin(__builtin_amdgcn_s_wait_tensorcnt)
#define HAVE_TDM 1
#endif
#endif
#ifndef HAVE_TDM
#define HAVE_TDM 0
#endif

typedef __attribute__((ext_vector_type(4))) unsigned int ui4;
typedef __attribute__((ext_vector_type(8))) int si8;
typedef __attribute__((ext_vector_type(4))) int si4;

__device__ __forceinline__ u16_t f2bf(float f) {
  unsigned u = __float_as_uint(f);
  u += 0x7FFFu + ((u >> 16) & 1u);    // round-to-nearest-even
  return (u16_t)(u >> 16);
}

__device__ __forceinline__ v8f zero8() {
  v8f z;
#pragma unroll
  for (int i = 0; i < 8; ++i) z[i] = 0.0f;
  return z;
}

__device__ __forceinline__ v16bf make_frag(const u16_t* p0, const u16_t* p1) {
  union { v16bf v; v8bf h[2]; } u;
  u.h[0] = *reinterpret_cast<const v8bf*>(p0);
  u.h[1] = *reinterpret_cast<const v8bf*>(p1);
  return u.v;
}

// A-frag (16x32 bf16): rowk points at [row][k_chunk_base], row-major, k contiguous.
// ISA layout: VGPR0-3 hold k = 8*half + 0..7 ; VGPR4-7 hold k = 16 + 8*half + 0..7
__device__ __forceinline__ v16bf load_a(const u16_t* rowk, int hl) {
  return make_frag(rowk + 8 * hl, rowk + 16 + 8 * hl);
}

// B-frag (32x16 bf16): colk points at [col][k_chunk_base] in k-contiguous (transposed)
// storage. ISA layout: lane holds col = lane&15, k = 8*half + 0..7 then 16 + 8*half + 0..7.
__device__ __forceinline__ v16bf load_b(const u16_t* colk, int hl) {
  return make_frag(colk + 8 * hl, colk + 16 + 8 * hl);
}

__device__ __forceinline__ v8f wmma_bf16(v16bf a, v16bf b, v8f c) {
  return __builtin_amdgcn_wmma_f32_16x16x32_bf16(false, a, false, b, (short)0, c,
                                                 false, false);
}

#if HAVE_TDM
// 2-D TDM copy: tile_rows x tile_k bf16 tile, k contiguous, row stride in elements.
// D# built per CDNA5 ISA 8.3/8.4 (group0: count/lds/global/type, group1: dims).
__device__ __forceinline__ void tdm_load_tile_bf16(unsigned lds_off,
                                                   const u16_t* gsrc,
                                                   unsigned tile_k, unsigned tile_rows,
                                                   unsigned tensor_k, unsigned tensor_rows,
                                                   unsigned long long row_stride) {
  unsigned long long ga = (unsigned long long)(uintptr_t)(const void*)gsrc;
  ui4 g0;
  g0[0] = 1u;                                              // count=1, user descriptor
  g0[1] = lds_off;                                         // LDS byte address
  g0[2] = (unsigned)(ga & 0xFFFFFFFFull);                  // global_addr[31:0]
  g0[3] = (unsigned)((ga >> 32) & 0x01FFFFFFull) | (2u << 30); // addr[56:32] | type=2
  si8 g1;
  g1[0] = (int)(1u << 16);                                 // data_size=1 (2B); mask=0
  g1[1] = (int)((tensor_k & 0xFFFFu) << 16);               // tensor_dim0[15:0]
  g1[2] = (int)((tensor_k >> 16) | ((tensor_rows & 0xFFFFu) << 16)); // td0 hi|td1 lo
  g1[3] = (int)((tensor_rows >> 16) | ((tile_k & 0xFFFFu) << 16));   // td1 hi|tile_dim0
  g1[4] = (int)(tile_rows & 0xFFFFu);                      // tile_dim1; tile_dim2=0
  g1[5] = (int)(unsigned)(row_stride & 0xFFFFFFFFull);     // tensor_dim0_stride lo32
  g1[6] = (int)(unsigned)((row_stride >> 32) & 0xFFFFull); // stride hi16; td1_stride=0
  g1[7] = 0;
  si4 gz; gz[0] = 0; gz[1] = 0; gz[2] = 0; gz[3] = 0;      // 2-D: groups 2/3 unused
#if __clang_major__ >= 23
  si8 z8; z8[0]=0; z8[1]=0; z8[2]=0; z8[3]=0; z8[4]=0; z8[5]=0; z8[6]=0; z8[7]=0;
  __builtin_amdgcn_tensor_load_to_lds(g0, g1, gz, gz, z8, 0);
#else
  __builtin_amdgcn_tensor_load_to_lds(g0, g1, gz, gz, 0);
#endif
}
#else
// Fallback: cooperative global->LDS copy of 128 rows x 32 bf16 (row stride D_).
__device__ __forceinline__ void stage_fallback(u16_t* dst, const u16_t* src, int tid) {
#pragma unroll
  for (int it = 0; it < 8; ++it) {
    int idx = tid + it * 128;            // 1024 x u64 total
    int n = idx >> 3, q = idx & 7;
    reinterpret_cast<unsigned long long*>(dst + n * 32)[q] =
        reinterpret_cast<const unsigned long long*>(src + (size_t)n * D_)[q];
  }
}
#endif

// ---------------- conversion / transpose kernels ----------------
__global__ void k_cvt_x(const float* __restrict__ x, u16_t* __restrict__ xb) {
  int i = blockIdx.x * 256 + threadIdx.x;            // grid sized exactly
  xb[i] = f2bf(x[i]);
}

struct WPtrs { const float* p[9]; };

// Wt[(w9*H+h)][n][k] = W[w9][h][k][n]   (w9 = group*3 + {q,k,v})
__global__ void k_cvt_w(WPtrs wp, u16_t* __restrict__ Wt) {
  int i = blockIdx.x * 256 + threadIdx.x;
  int k = i & (D_ - 1);
  int r = i >> 10;
  int n = r & (DH_ - 1);
  r >>= 7;
  int h  = r & 1;
  int w9 = r >> 1;
  Wt[i] = f2bf(wp.p[w9][((size_t)h * D_ + k) * DH_ + n]);
}

// WoT[n][k] = W_o[k][n]
__global__ void k_cvt_wo(const float* __restrict__ Wo, u16_t* __restrict__ WoT) {
  int i = blockIdx.x * 256 + threadIdx.x;
  int k = i & (DH_ - 1);
  int n = i >> 7;
  WoT[i] = f2bf(Wo[(size_t)k * D_ + n]);
}

// ---------------- QKV projection GEMM ----------------
// One wave: 16 rows x 128 cols of one (group, q/k/v, head) projection.
// All 4 waves of a block share one weight matrix; its 128x32 k-chunks are
// staged into double-buffered LDS by the Tensor Data Mover and consumed as
// B-fragments via ds_load_b128. jobs = 18 mats * 512 Mtiles = 9216 waves.
__global__ __launch_bounds__(128) void k_qkv(const u16_t* __restrict__ xb,
                                             const u16_t* __restrict__ Wt,
                                             u16_t* __restrict__ Q,
                                             u16_t* __restrict__ K,
                                             u16_t* __restrict__ Vt) {
  const int wave = threadIdx.x >> 5;
  const int lane = threadIdx.x & 31;
  const int hl = lane >> 4, cl = lane & 15;
  const int job = blockIdx.x * 4 + wave;
  const int mt  = job & 511;
  const int mat = job >> 9;            // 0..17 (uniform across the block)
  const int h   = mat & 1;
  const int w9  = mat >> 1;            // 0..8
  const int g   = w9 / 3;
  const int j   = w9 % 3;              // 0=q 1=k 2=v
  const int mbase = mt * 16;

  __shared__ __align__(16) u16_t sW[2][128 * 32];   // double-buffered weight chunks

  const u16_t* arow = xb + (size_t)(mbase + cl) * D_;
  const u16_t* wmat = Wt + (size_t)(w9 * H_ + h) * (size_t)(DH_ * D_);

  v8f acc[8];
#pragma unroll
  for (int nt = 0; nt < 8; ++nt) acc[nt] = zero8();

  // stage k-chunk 0
#if HAVE_TDM
  if (wave == 0) {
    tdm_load_tile_bf16((unsigned)(uintptr_t)(void*)&sW[0][0], wmat,
                       32u, 128u, (unsigned)D_, (unsigned)DH_,
                       (unsigned long long)D_);
    __builtin_amdgcn_s_wait_tensorcnt(0);
  }
#else
  stage_fallback(&sW[0][0], wmat, threadIdx.x);
#endif
  __syncthreads();

  for (int kc = 0; kc < 32; ++kc) {
    const u16_t* wl = &sW[kc & 1][0];
    // kick off next chunk while computing this one
#if HAVE_TDM
    if (wave == 0 && kc + 1 < 32)
      tdm_load_tile_bf16((unsigned)(uintptr_t)(void*)&sW[(kc + 1) & 1][0],
                         wmat + 32 * (kc + 1), 32u, 128u,
                         (unsigned)D_, (unsigned)DH_, (unsigned long long)D_);
#else
    if (kc + 1 < 32) stage_fallback(&sW[(kc + 1) & 1][0], wmat + 32 * (kc + 1),
                                    threadIdx.x);
#endif
    v16bf a = load_a(arow + 32 * kc, hl);
    if (kc + 1 < 32) __builtin_prefetch(arow + 32 * (kc + 1), 0, 1);
    // batch all 8 B-fragment loads into one clause, then a pure WMMA chain
    v16bf bfr[8];
#pragma unroll
    for (int nt = 0; nt < 8; ++nt)
      bfr[nt] = load_b(wl + (nt * 16 + cl) * 32, hl);
#pragma unroll
    for (int nt = 0; nt < 8; ++nt)
      acc[nt] = wmma_bf16(a, bfr[nt], acc[nt]);
#if HAVE_TDM
    if (wave == 0 && kc + 1 < 32) __builtin_amdgcn_s_wait_tensorcnt(0);
#endif
    __syncthreads();
  }

  const int b = mbase >> 11;           // / T_
  const int t = mbase & (T_ - 1);
  const size_t base = (size_t)((g * H_ + h) * B_ + b) * (size_t)(T_ * DH_);
#pragma unroll
  for (int nt = 0; nt < 8; ++nt) {
#pragma unroll
    for (int r = 0; r < 8; ++r) {
      const int   tr = t + r + 8 * hl;
      const int   n  = nt * 16 + cl;
      const u16_t v  = f2bf(acc[nt][r]);
      if (j == 0)      Q[base + (size_t)tr * DH_ + n] = v;       // row-major [T][DH]
      else if (j == 1) K[base + (size_t)tr * DH_ + n] = v;       // row-major [T][DH]
      else             Vt[base + (size_t)n * T_ + tr] = v;       // transposed [DH][T]
    }
  }
}

// ---------------- flash attention (both heads averaged) ----------------
// One wave per (b, group, 16-row q tile); jobs = 4*3*128 = 1536 waves.
__global__ __launch_bounds__(128) void k_attn(const u16_t* __restrict__ Q,
                                              const u16_t* __restrict__ K,
                                              const u16_t* __restrict__ Vt,
                                              const float* __restrict__ rel_emb,
                                              float* __restrict__ head_outs) {
  const int wave = threadIdx.x >> 5;
  const int lane = threadIdx.x & 31;
  const int hl = lane >> 4, cl = lane & 15;
  const int job = blockIdx.x * 4 + wave;
  const int qt  = job & 127;
  const int tmp = job >> 7;
  const int b   = tmp & 3;
  const int g   = tmp >> 2;            // 0=local 1=global 2=rel
  const int t_base = qt * 16;
  const float inv_scale = 0.0883883476483184f;  // 1/sqrt(128)

  __shared__ __align__(16) u16_t sP[4][16 * 32];  // per-wave P staging
  u16_t* myP = sP[wave];

  float fin[8][8];
#pragma unroll
  for (int dt = 0; dt < 8; ++dt)
#pragma unroll
    for (int r = 0; r < 8; ++r) fin[dt][r] = 0.0f;

  for (int h = 0; h < H_; ++h) {
    const size_t hb = (size_t)((g * H_ + h) * B_ + b) * (size_t)(T_ * DH_);
    const u16_t* qb = Q + hb;
    const u16_t* kb = K + hb;
    const u16_t* vb = Vt + hb;

    v16bf qf[4];
    const u16_t* qrow = qb + (size_t)(t_base + cl) * DH_;
#pragma unroll
    for (int kc = 0; kc < 4; ++kc) qf[kc] = load_a(qrow + 32 * kc, hl);

    float m[8], l[8];
    v8f acc[8];
#pragma unroll
    for (int r = 0; r < 8; ++r) { m[r] = -1.0e30f; l[r] = 0.0f; }
#pragma unroll
    for (int dt = 0; dt < 8; ++dt) acc[dt] = zero8();

    int sc_lo = 0;
    if (g == 0) { int raw = t_base - WIN_; sc_lo = raw > 0 ? (raw & ~31) : 0; }

    for (int sc = sc_lo; sc <= t_base + 15; sc += 32) {
      // ---- scores: two 16x16 tiles over DH=128; batch all K-frag loads,
      //      then interleave the two accumulation chains ----
      v16bf kf[8];
#pragma unroll
      for (int kc = 0; kc < 4; ++kc)
        kf[kc] = load_b(kb + (size_t)(sc + cl) * DH_ + 32 * kc, hl);
#pragma unroll
      for (int kc = 0; kc < 4; ++kc)
        kf[4 + kc] = load_b(kb + (size_t)(sc + 16 + cl) * DH_ + 32 * kc, hl);
      v8f s0 = zero8(), s1 = zero8();
#pragma unroll
      for (int kc = 0; kc < 4; ++kc) {
        s0 = wmma_bf16(qf[kc], kf[kc], s0);
        s1 = wmma_bf16(qf[kc], kf[4 + kc], s1);
      }
      // ---- online softmax over the 32-wide chunk ----
#pragma unroll
      for (int r = 0; r < 8; ++r) {
        const int t  = t_base + r + 8 * hl;
        const int sa = sc + cl;
        const int sb = sc + 16 + cl;
        float v0 = s0[r] * inv_scale;
        float v1 = s1[r] * inv_scale;
        if (g == 2) {
          int d0 = sa - t; d0 = d0 < -MRP_ ? -MRP_ : (d0 > MRP_ ? MRP_ : d0);
          int d1 = sb - t; d1 = d1 < -MRP_ ? -MRP_ : (d1 > MRP_ ? MRP_ : d1);
          v0 += rel_emb[h * RELV_ + d0 + MRP_];
          v1 += rel_emb[h * RELV_ + d1 + MRP_];
        }
        const bool k0 = (sa > t) || (g == 0 && sa < t - WIN_);
        const bool k1 = (sb > t) || (g == 0 && sb < t - WIN_);
        v0 = k0 ? -1.0e30f : v0;
        v1 = k1 ? -1.0e30f : v1;
        float mx = fmaxf(v0, v1);
        mx = fmaxf(mx, __shfl_xor(mx, 1, 32));
        mx = fmaxf(mx, __shfl_xor(mx, 2, 32));
        mx = fmaxf(mx, __shfl_xor(mx, 4, 32));
        mx = fmaxf(mx, __shfl_xor(mx, 8, 32));
        const float mn   = fmaxf(m[r], mx);
        const float corr = __expf(m[r] - mn);
        const float p0   = __expf(v0 - mn);
        const float p1   = __expf(v1 - mn);
        float rs = p0 + p1;
        rs += __shfl_xor(rs, 1, 32);
        rs += __shfl_xor(rs, 2, 32);
        rs += __shfl_xor(rs, 4, 32);
        rs += __shfl_xor(rs, 8, 32);
        l[r] = l[r] * corr + rs;
        m[r] = mn;
#pragma unroll
        for (int dt = 0; dt < 8; ++dt) acc[dt][r] *= corr;
        myP[(r + 8 * hl) * 32 + cl]      = f2bf(p0);
        myP[(r + 8 * hl) * 32 + 16 + cl] = f2bf(p1);
      }
      asm volatile("s_wait_dscnt 0x0" ::: "memory");   // in-wave LDS RAW
      // ---- P @ V over this chunk: batch V-frag loads, then WMMA chain ----
      v16bf pf = load_a(myP + cl * 32, hl);
      v16bf vf[8];
#pragma unroll
      for (int dt = 0; dt < 8; ++dt)
        vf[dt] = load_b(vb + (size_t)(dt * 16 + cl) * T_ + sc, hl);
#pragma unroll
      for (int dt = 0; dt < 8; ++dt)
        acc[dt] = wmma_bf16(pf, vf[dt], acc[dt]);
    }
#pragma unroll
    for (int r = 0; r < 8; ++r) {
      const float s = 0.5f / l[r];                     // head mean
#pragma unroll
      for (int dt = 0; dt < 8; ++dt) fin[dt][r] += acc[dt][r] * s;
    }
  }

  float* ob = head_outs + (size_t)(g * B_ + b) * (size_t)(T_ * DH_);
#pragma unroll
  for (int dt = 0; dt < 8; ++dt)
#pragma unroll
    for (int r = 0; r < 8; ++r)
      ob[(size_t)(t_base + r + 8 * hl) * DH_ + dt * 16 + cl] = fin[dt][r];
}

// ---------------- router + top-2 combine ----------------
// One wave per token; 8192 waves.
__global__ __launch_bounds__(256) void k_router(const float* __restrict__ x,
                                                const float* __restrict__ rW,
                                                const float* __restrict__ rb,
                                                const float* __restrict__ head_outs,
                                                u16_t* __restrict__ comb) {
  const int wave = threadIdx.x >> 5;
  const int lane = threadIdx.x & 31;
  const int tok = blockIdx.x * 8 + wave;
  const float* xr = x + (size_t)tok * D_;
  float l0 = 0.f, l1 = 0.f, l2 = 0.f;
  for (int d = lane; d < D_; d += 32) {
    const float xv = xr[d];
    l0 = fmaf(xv, rW[d * 3 + 0], l0);
    l1 = fmaf(xv, rW[d * 3 + 1], l1);
    l2 = fmaf(xv, rW[d * 3 + 2], l2);
  }
#pragma unroll
  for (int msk = 16; msk >= 1; msk >>= 1) {
    l0 += __shfl_xor(l0, msk, 32);
    l1 += __shfl_xor(l1, msk, 32);
    l2 += __shfl_xor(l2, msk, 32);
  }
  l0 += rb[0]; l1 += rb[1]; l2 += rb[2];
  const float lg[3] = {l0, l1, l2};
  int i0 = 0; float v0 = lg[0];
  if (lg[1] > v0) { v0 = lg[1]; i0 = 1; }
  if (lg[2] > v0) { v0 = lg[2]; i0 = 2; }
  int i1 = -1; float v1 = 0.f;
#pragma unroll
  for (int n = 0; n < 3; ++n) {
    if (n == i0) continue;
    if (i1 < 0 || lg[n] > v1) { v1 = lg[n]; i1 = n; }
  }
  const float e1  = __expf(v1 - v0);
  const float inv = 1.0f / (1.0f + e1);
  const float g0 = inv, g1 = e1 * inv;
  const float* h0 = head_outs + (size_t)i0 * (size_t)(BT_ * DH_) + (size_t)tok * DH_;
  const float* h1 = head_outs + (size_t)i1 * (size_t)(BT_ * DH_) + (size_t)tok * DH_;
#pragma unroll
  for (int d0 = 0; d0 < DH_; d0 += 32) {
    const int d = d0 + lane;
    comb[(size_t)tok * DH_ + d] = f2bf(g0 * h0[d] + g1 * h1[d]);
  }
}

// ---------------- output projection GEMM + bias ----------------
// One wave: 16 x 64 output tile; jobs = 512 * 16 = 8192 waves.
__global__ __launch_bounds__(128) void k_outproj(const u16_t* __restrict__ comb,
                                                 const u16_t* __restrict__ WoT,
                                                 const float* __restrict__ bias,
                                                 float* __restrict__ out) {
  const int wave = threadIdx.x >> 5;
  const int lane = threadIdx.x & 31;
  const int hl = lane >> 4, cl = lane & 15;
  const int job = blockIdx.x * 4 + wave;
  const int mt = job & 511;
  const int ng = job >> 9;             // 0..15
  const int mbase = mt * 16;
  const int nbase = ng * 64;
  const u16_t* arow = comb + (size_t)(mbase + cl) * DH_;
  v8f acc[4];
#pragma unroll
  for (int nt = 0; nt < 4; ++nt) acc[nt] = zero8();
#pragma unroll
  for (int kc = 0; kc < 4; ++kc) {
    v16bf a = load_a(arow + 32 * kc, hl);
    v16bf bfr[4];
#pragma unroll
    for (int nt = 0; nt < 4; ++nt)
      bfr[nt] = load_b(WoT + (size_t)(nbase + nt * 16 + cl) * DH_ + 32 * kc, hl);
#pragma unroll
    for (int nt = 0; nt < 4; ++nt)
      acc[nt] = wmma_bf16(a, bfr[nt], acc[nt]);
  }
#pragma unroll
  for (int nt = 0; nt < 4; ++nt) {
    const int   n  = nbase + nt * 16 + cl;
    const float bn = bias[n];
#pragma unroll
    for (int r = 0; r < 8; ++r)
      out[(size_t)(mbase + r + 8 * hl) * D_ + n] = acc[nt][r] + bn;
  }
}

// ---------------- host launch ----------------
extern "C" void kernel_launch(void* const* d_in, const int* in_sizes, int n_in,
                              void* d_out, int out_size, void* d_ws, size_t ws_size,
                              hipStream_t stream) {
  (void)in_sizes; (void)n_in; (void)out_size; (void)ws_size;
  const float* x        = (const float*)d_in[0];
  const float* rel_emb  = (const float*)d_in[10];
  const float* router_W = (const float*)d_in[11];
  const float* router_b = (const float*)d_in[12];
  const float* W_o      = (const float*)d_in[13];
  const float* W_o_b    = (const float*)d_in[14];
  float* out = (float*)d_out;

  char* ws = (char*)d_ws;
  size_t off = 0;
  auto take = [&](size_t bytes) -> char* {
    char* p = ws + off;
    off = (off + bytes + 255) & ~(size_t)255;
    return p;
  };
  u16_t* xb   = (u16_t*)take((size_t)BT_ * D_ * 2);                 // 16.8 MB
  u16_t* Wt   = (u16_t*)take((size_t)9 * H_ * DH_ * D_ * 2);        //  4.7 MB
  u16_t* WoT  = (u16_t*)take((size_t)D_ * DH_ * 2);                 //  0.3 MB
  u16_t* Qb   = (u16_t*)take((size_t)G_ * H_ * BT_ * DH_ * 2);      // 12.6 MB
  u16_t* Kb   = (u16_t*)take((size_t)G_ * H_ * BT_ * DH_ * 2);      // 12.6 MB
  u16_t* Vtb  = (u16_t*)take((size_t)G_ * H_ * BT_ * DH_ * 2);      // 12.6 MB
  float* head = (float*)take((size_t)G_ * BT_ * DH_ * 4);           // 12.6 MB
  u16_t* comb = (u16_t*)take((size_t)BT_ * DH_ * 2);                //  2.1 MB

  WPtrs wp;
  for (int i = 0; i < 9; ++i) wp.p[i] = (const float*)d_in[1 + i];

  k_cvt_x  <<<(BT_ * D_) / 256, 256, 0, stream>>>(x, xb);
  k_cvt_w  <<<(9 * H_ * DH_ * D_) / 256, 256, 0, stream>>>(wp, Wt);
  k_cvt_wo <<<(D_ * DH_) / 256, 256, 0, stream>>>(W_o, WoT);
  k_qkv    <<<2304, 128, 0, stream>>>(xb, Wt, Qb, Kb, Vtb);         // 9216 waves
  k_attn   <<<384, 128, 0, stream>>>(Qb, Kb, Vtb, rel_emb, head);   // 1536 waves
  k_router <<<1024, 256, 0, stream>>>(x, router_W, router_b, head, comb);
  k_outproj<<<2048, 128, 0, stream>>>(comb, WoT, W_o_b, out);       // 8192 waves
}
